// DecoderLayer_15144054686433
// MI455X (gfx1250) — matrix operand order repaired and checked by reading, not compiled, not run on hardware
//
#include <hip/hip_runtime.h>

typedef unsigned short u16;
typedef __attribute__((ext_vector_type(16))) __bf16 v16bf;
typedef __attribute__((ext_vector_type(8)))  float  v8f;
typedef __attribute__((ext_vector_type(4)))  int    v4i;

#if defined(__gfx1250__) && __has_builtin(__builtin_amdgcn_global_load_async_to_lds_b128) && __has_builtin(__builtin_amdgcn_s_wait_asynccnt)
#define USE_ASYNC 1
typedef __attribute__((address_space(1))) v4i gv4i;   // global int4 (AS1)
typedef __attribute__((address_space(3))) v4i lv4i;   // LDS int4 (AS3)
#define AS1(p) ((gv4i*)(p))
#define AS3(p) ((lv4i*)(p))
#else
#define USE_ASYNC 0
#endif

__device__ __forceinline__ u16 f2bf(float f) {
  unsigned x = __float_as_uint(f);
  unsigned r = (x + 0x7fffu + ((x >> 16) & 1u)) >> 16;   // round-to-nearest-even
  return (u16)r;
}

union FragBF { v16bf v; unsigned u[8]; };

// ---------------------------------------------------------------------------
// Generic bf16 GEMM:  C[M,N] = A[M,K] * Bst[N,K]^T   (Bst rows are N-vectors
// over K, i.e. exactly torch Linear weight layout [out,in]).
// Block tile 128x128, K-stage 64 == 16x v_wmma_f32_16x16x32_bf16 per stage.
// Double-buffered LDS staging via GLOBAL_LOAD_ASYNC_TO_LDS_B128 (ASYNCcnt):
// while stage s computes, stage s+1 streams straight into the other buffer.
// 8 waves: 2 (M) x 4 (N); each wave computes a 64x32 subtile (4x2 wmma tiles).
// ---------------------------------------------------------------------------
#define BM 128
#define BN 128
#define BK 64
#define LDSS 72                  // padded LDS row stride (144B, 16B aligned)
#define BUFSZ (BM * LDSS)

__global__ __launch_bounds__(256) void gemm_bf16(
    const u16* __restrict__ A, const u16* __restrict__ B,
    const float* __restrict__ bias,
    float* __restrict__ Cf, u16* __restrict__ Cb,
    int M, int N, int K, int lda, int ldb, int ldc,
    long sA, long sB, long sC,
    float scale, int relu, int vtmode)
{
  __shared__ __align__(16) u16 As[2 * BUFSZ];
  __shared__ __align__(16) u16 Bs[2 * BUFSZ];

  const int z = blockIdx.z;
  A += (long)z * sA;
  B += (long)z * sB;
  const long cbase = (long)z * sC;

  const int bm = blockIdx.y * BM;
  const int bn = blockIdx.x * BN;
  const int tid  = threadIdx.x;
  const int lane = tid & 31;
  const int wave = tid >> 5;
  const int r  = lane & 15;          // C column / A row / B row within tile
  const int hi = lane >> 4;          // lane half selector
  const int wm = (wave >> 2) * 64;   // wave M offset in block tile
  const int wn = (wave & 3) * 32;    // wave N offset in block tile

  v8f acc[4][2] = {};

  const int srow = tid >> 1;         // staging row 0..127
  const int scol = (tid & 1) * 32;   // staging col 0 or 32 (elems)

  // issue one K-stage of staging (A: always full; B: guarded for N=64 case)
  auto issue = [&](int buf, int k0) {
    const u16* ga = A + (size_t)(bm + srow) * lda + k0 + scol;
    u16* la = As + buf * BUFSZ + srow * LDSS + scol;
    const int bok = (bn + srow < N);
    const u16* gb = B + (size_t)(bn + srow) * ldb + k0 + scol;
    u16* lb = Bs + buf * BUFSZ + srow * LDSS + scol;
#if USE_ASYNC
    __builtin_amdgcn_global_load_async_to_lds_b128(AS1(ga), AS3(la),  0, 0);
    __builtin_amdgcn_global_load_async_to_lds_b128(AS1(ga), AS3(la), 16, 0);
    __builtin_amdgcn_global_load_async_to_lds_b128(AS1(ga), AS3(la), 32, 0);
    __builtin_amdgcn_global_load_async_to_lds_b128(AS1(ga), AS3(la), 48, 0);
    if (bok) {
      __builtin_amdgcn_global_load_async_to_lds_b128(AS1(gb), AS3(lb),  0, 0);
      __builtin_amdgcn_global_load_async_to_lds_b128(AS1(gb), AS3(lb), 16, 0);
      __builtin_amdgcn_global_load_async_to_lds_b128(AS1(gb), AS3(lb), 32, 0);
      __builtin_amdgcn_global_load_async_to_lds_b128(AS1(gb), AS3(lb), 48, 0);
    }
    // unissued B rows (n >= N) leave stale-but-finite bf16 in LDS; those
    // accumulator columns are never stored in the epilogue.
#else
    {
      const uint4* s4 = (const uint4*)ga;
      uint4 t0 = s4[0], t1 = s4[1], t2 = s4[2], t3 = s4[3];
      uint4* d4 = (uint4*)la;
      d4[0] = t0; d4[1] = t1; d4[2] = t2; d4[3] = t3;
    }
    {
      uint4 t0 = {0,0,0,0}, t1 = {0,0,0,0}, t2 = {0,0,0,0}, t3 = {0,0,0,0};
      if (bok) {
        const uint4* s4 = (const uint4*)gb;
        t0 = s4[0]; t1 = s4[1]; t2 = s4[2]; t3 = s4[3];
      }
      uint4* d4 = (uint4*)lb;
      d4[0] = t0; d4[1] = t1; d4[2] = t2; d4[3] = t3;
    }
#endif
  };

  const int nstg = K / BK;
  issue(0, 0);

  for (int s = 0; s < nstg; ++s) {
    const int cur = s & 1;
#if USE_ASYNC
    __builtin_amdgcn_s_wait_asynccnt(0);   // our stage-`cur` LDS writes landed
#endif
    __syncthreads();                       // everyone's landed; prev reads done
    if (s + 1 < nstg) issue(cur ^ 1, (s + 1) * BK);  // overlap with compute

    const u16* Ab = As + cur * BUFSZ;
    const u16* Bb = Bs + cur * BUFSZ;
#pragma unroll
    for (int ks = 0; ks < BK; ks += 32) {
      // ---- gather fragments per ISA 7.12.2 layouts (lower to ds_load_b128)
      FragBF af[4], bf2[2];
#pragma unroll
      for (int mt = 0; mt < 4; ++mt) {
        const u16* ap = Ab + (wm + mt * 16 + r) * LDSS + ks;
#pragma unroll
        for (int p = 0; p < 8; ++p) {
          int kk = hi * 8 + (p < 4 ? 2 * p : 2 * p + 8);   // A 16x32 bf16
          af[mt].u[p] = *(const unsigned*)(ap + kk);
        }
      }
#pragma unroll
      for (int nt = 0; nt < 2; ++nt) {
        const u16* bp = Bb + (wn + nt * 16 + r) * LDSS + ks;
#pragma unroll
        for (int p = 0; p < 8; ++p) {
          int kk = hi * 8 + (p < 4 ? 2 * p : 2 * p + 8);   // B 32x16 bf16
          bf2[nt].u[p] = *(const unsigned*)(bp + kk);
        }
      }
#pragma unroll
      for (int mt = 0; mt < 4; ++mt)
#pragma unroll
        for (int nt = 0; nt < 2; ++nt)
          acc[mt][nt] = __builtin_amdgcn_wmma_f32_16x16x32_bf16(
              false, af[mt].v, false, bf2[nt].v, (short)0, acc[mt][nt],
              false, false);
    }
    __syncthreads();                       // stage `cur` free for reuse
  }

  // ---- epilogue: C layout row = vgpr + 8*hi, col = r
#pragma unroll
  for (int mt = 0; mt < 4; ++mt) {
#pragma unroll
    for (int nt = 0; nt < 2; ++nt) {
      const int n = bn + wn + nt * 16 + r;
      if (n >= N) continue;
      const float bv = bias ? bias[n] : 0.0f;
#pragma unroll
      for (int j = 0; j < 8; ++j) {
        const int m = bm + wm + mt * 16 + hi * 8 + j;
        float val = acc[mt][nt][j] * scale + bv;
        if (relu) val = fmaxf(val, 0.0f);
        if (vtmode) {
          // store V transposed per head: vt[((b*16+h)*64+d)*1024 + t]
          int b = m >> 10, t = m & 1023, h = n >> 6, d = n & 63;
          Cb[((long)((b * 16 + h) * 64 + d) << 10) + t] = f2bf(val);
        } else if (Cb) {
          Cb[cbase + (long)m * ldc + n] = f2bf(val);
        } else {
          Cf[cbase + (long)m * ldc + n] = val;
        }
      }
    }
  }
}

// ---------------------------------------------------------------------------
// Row softmax over `cols` f32 -> bf16 probabilities. One block (8 waves) per row.
// ---------------------------------------------------------------------------
__global__ __launch_bounds__(256) void softmax_row(
    const float* __restrict__ S, u16* __restrict__ P, int cols)
{
  const long row = blockIdx.x;
  const float* s = S + row * (long)cols;
  u16* p = P + row * (long)cols;
  const int tid = threadIdx.x;
  __shared__ float red[8];

  float m = -3.402823466e38f;
  for (int c = tid; c < cols; c += 256) m = fmaxf(m, s[c]);
#pragma unroll
  for (int o = 16; o > 0; o >>= 1) m = fmaxf(m, __shfl_xor(m, o, 32));
  if ((tid & 31) == 0) red[tid >> 5] = m;
  __syncthreads();
#pragma unroll
  for (int i = 0; i < 8; ++i) m = fmaxf(m, red[i]);
  __syncthreads();

  float sum = 0.0f;
  for (int c = tid; c < cols; c += 256) sum += __expf(s[c] - m);
#pragma unroll
  for (int o = 16; o > 0; o >>= 1) sum += __shfl_xor(sum, o, 32);
  if ((tid & 31) == 0) red[tid >> 5] = sum;
  __syncthreads();
  float tot = 0.0f;
#pragma unroll
  for (int i = 0; i < 8; ++i) tot += red[i];
  const float rinv = 1.0f / tot;

  for (int c = tid; c < cols; c += 256) p[c] = f2bf(__expf(s[c] - m) * rinv);
}

// ---------------------------------------------------------------------------
// y = LayerNorm(x + res) * g + b  over D=1024; writes f32 (and optional bf16).
// One block per row, 4 elems/thread.
// ---------------------------------------------------------------------------
__global__ __launch_bounds__(256) void add_ln(
    const float* __restrict__ X, const float* __restrict__ R,
    const float* __restrict__ g, const float* __restrict__ be,
    float* __restrict__ Y, u16* __restrict__ Yb)
{
  const long row = blockIdx.x;
  const float* xr = X + row * 1024;
  const float* rr = R + row * 1024;
  const int tid = threadIdx.x;
  __shared__ float red[8];

  float v[4]; float s = 0.0f;
#pragma unroll
  for (int i = 0; i < 4; ++i) { v[i] = xr[tid + 256 * i] + rr[tid + 256 * i]; s += v[i]; }
#pragma unroll
  for (int o = 16; o > 0; o >>= 1) s += __shfl_xor(s, o, 32);
  if ((tid & 31) == 0) red[tid >> 5] = s;
  __syncthreads();
  float tot = 0.0f;
#pragma unroll
  for (int i = 0; i < 8; ++i) tot += red[i];
  const float mean = tot * (1.0f / 1024.0f);
  __syncthreads();

  float q = 0.0f;
#pragma unroll
  for (int i = 0; i < 4; ++i) { float d = v[i] - mean; q += d * d; }
#pragma unroll
  for (int o = 16; o > 0; o >>= 1) q += __shfl_xor(q, o, 32);
  if ((tid & 31) == 0) red[tid >> 5] = q;
  __syncthreads();
  float qt = 0.0f;
#pragma unroll
  for (int i = 0; i < 8; ++i) qt += red[i];
  const float rstd = rsqrtf(qt * (1.0f / 1024.0f) + 1e-5f);

#pragma unroll
  for (int i = 0; i < 4; ++i) {
    const int c = tid + 256 * i;
    const float o = (v[i] - mean) * rstd * g[c] + be[c];
    Y[row * 1024 + c] = o;
    if (Yb) Yb[row * 1024 + c] = f2bf(o);
  }
}

// ---------------------------------------------------------------------------
// f32 -> bf16 conversion (grid-stride)
// ---------------------------------------------------------------------------
__global__ __launch_bounds__(256) void cvt_kernel(
    const float* __restrict__ in, u16* __restrict__ out, long n)
{
  long i = (long)blockIdx.x * 256 + threadIdx.x;
  const long stride = (long)gridDim.x * 256;
  for (; i < n; i += stride) out[i] = f2bf(in[i]);
}

// ---------------------------------------------------------------------------
extern "C" void kernel_launch(void* const* d_in, const int* in_sizes, int n_in,
                              void* d_out, int out_size, void* d_ws, size_t ws_size,
                              hipStream_t stream) {
  (void)in_sizes; (void)n_in; (void)out_size; (void)ws_size;

  const float* x   = (const float*)d_in[0];
  const float* mem = (const float*)d_in[1];
  const float* Wq1 = (const float*)d_in[2];  const float* bq1 = (const float*)d_in[3];
  const float* Wk1 = (const float*)d_in[4];  const float* bk1 = (const float*)d_in[5];
  const float* Wv1 = (const float*)d_in[6];  const float* bv1 = (const float*)d_in[7];
  const float* Wq2 = (const float*)d_in[8];  const float* bq2 = (const float*)d_in[9];
  const float* Wk2 = (const float*)d_in[10]; const float* bk2 = (const float*)d_in[11];
  const float* Wv2 = (const float*)d_in[12]; const float* bv2 = (const float*)d_in[13];
  const float* W1  = (const float*)d_in[14]; const float* b1  = (const float*)d_in[15];
  const float* W2  = (const float*)d_in[16]; const float* b2  = (const float*)d_in[17];
  const float* g1  = (const float*)d_in[18]; const float* be1 = (const float*)d_in[19];
  const float* g2  = (const float*)d_in[20]; const float* be2 = (const float*)d_in[21];
  const float* g3  = (const float*)d_in[22]; const float* be3 = (const float*)d_in[23];
  float* out = (float*)d_out;

  const long TOK = 8 * 1024;   // B*N tokens
  const long DM  = 1024;
  const long FFD = 4096;
  const long SEQ = 1024;
  const int  HB  = 16;         // heads

  char* ws = (char*)d_ws;
  size_t off = 0;
  auto take = [&](size_t bytes) -> char* {
    char* p = ws + off;
    off += (bytes + 255) & ~(size_t)255;
    return p;
  };

  u16*   xb   = (u16*)take(TOK * DM * 2);
  u16*   memb = (u16*)take(TOK * DM * 2);
  u16*   wq1b = (u16*)take(DM * DM * 2);
  u16*   wk1b = (u16*)take(DM * DM * 2);
  u16*   wv1b = (u16*)take(DM * DM * 2);
  u16*   wq2b = (u16*)take(DM * DM * 2);
  u16*   wk2b = (u16*)take(DM * DM * 2);
  u16*   wv2b = (u16*)take(DM * DM * 2);
  u16*   w1b  = (u16*)take(FFD * DM * 2);
  u16*   w2b  = (u16*)take(DM * FFD * 2);
  u16*   qb   = (u16*)take(TOK * DM * 2);
  u16*   kb   = (u16*)take(TOK * DM * 2);
  u16*   vtb  = (u16*)take(TOK * DM * 2);
  float* Sws  = (float*)take((size_t)HB * SEQ * SEQ * 4);   // one batch of scores
  u16*   Pws  = (u16*)take((size_t)HB * SEQ * SEQ * 2);     // one batch of probs
  float* ctx  = (float*)take(TOK * DM * 4);                  // attn / ffn output
  float* x1   = (float*)take(TOK * DM * 4);
  u16*   x1b  = (u16*)take(TOK * DM * 2);
  float* x2   = (float*)take(TOK * DM * 4);
  u16*   x2b  = (u16*)take(TOK * DM * 2);
  u16*   hb   = (u16*)take(TOK * FFD * 2);                   // ffn hidden

  auto cvt = [&](const float* in, u16* o, long n) {
    long blocks = (n + 1023) / 1024; if (blocks > 16384) blocks = 16384;
    cvt_kernel<<<dim3((unsigned)blocks), dim3(256), 0, stream>>>(in, o, n);
  };
  auto gemm = [&](const u16* A, const u16* B, const float* bias,
                  float* Cf, u16* Cb, int M, int N, int K,
                  int lda, int ldb, int ldc,
                  long sA, long sB, long sC, int Z,
                  float scale, int relu, int vt) {
    dim3 grid((N + BN - 1) / BN, (M + BM - 1) / BM, Z);
    gemm_bf16<<<grid, dim3(256), 0, stream>>>(A, B, bias, Cf, Cb,
        M, N, K, lda, ldb, ldc, sA, sB, sC, scale, relu, vt);
  };

  // ---- precision downconvert (activations + weights)
  cvt(x,   xb,   TOK * DM);
  cvt(mem, memb, TOK * DM);
  cvt(Wq1, wq1b, DM * DM);  cvt(Wk1, wk1b, DM * DM);  cvt(Wv1, wv1b, DM * DM);
  cvt(Wq2, wq2b, DM * DM);  cvt(Wk2, wk2b, DM * DM);  cvt(Wv2, wv2b, DM * DM);
  cvt(W1,  w1b,  FFD * DM); cvt(W2,  w2b,  DM * FFD);

  const float iscl = 0.125f;  // 1/sqrt(64)

  // ================= self-attention =================
  gemm(xb, wq1b, bq1, nullptr, qb,  (int)TOK, (int)DM, (int)DM, 1024, 1024, 1024, 0, 0, 0, 1, 1.0f, 0, 0);
  gemm(xb, wk1b, bk1, nullptr, kb,  (int)TOK, (int)DM, (int)DM, 1024, 1024, 1024, 0, 0, 0, 1, 1.0f, 0, 0);
  gemm(xb, wv1b, bv1, nullptr, vtb, (int)TOK, (int)DM, (int)DM, 1024, 1024, 1024, 0, 0, 0, 1, 1.0f, 0, 1);
  for (int b = 0; b < 8; ++b) {
    gemm(qb + (long)b * SEQ * DM, kb + (long)b * SEQ * DM, nullptr, Sws, nullptr,
         (int)SEQ, (int)SEQ, 64, 1024, 1024, 1024, 64, 64, SEQ * SEQ, HB, iscl, 0, 0);
    softmax_row<<<dim3(HB * (unsigned)SEQ), dim3(256), 0, stream>>>(Sws, Pws, (int)SEQ);
    gemm(Pws, vtb + (long)b * HB * 64 * SEQ, nullptr, ctx + (long)b * SEQ * DM, nullptr,
         (int)SEQ, 64, (int)SEQ, 1024, 1024, 1024, SEQ * SEQ, 64 * SEQ, 64, HB, 1.0f, 0, 0);
  }
  add_ln<<<dim3((unsigned)TOK), dim3(256), 0, stream>>>(x, ctx, g1, be1, x1, x1b);

  // ================= cross-attention =================
  gemm(x1b,  wq2b, bq2, nullptr, qb,  (int)TOK, (int)DM, (int)DM, 1024, 1024, 1024, 0, 0, 0, 1, 1.0f, 0, 0);
  gemm(memb, wk2b, bk2, nullptr, kb,  (int)TOK, (int)DM, (int)DM, 1024, 1024, 1024, 0, 0, 0, 1, 1.0f, 0, 0);
  gemm(memb, wv2b, bv2, nullptr, vtb, (int)TOK, (int)DM, (int)DM, 1024, 1024, 1024, 0, 0, 0, 1, 1.0f, 0, 1);
  for (int b = 0; b < 8; ++b) {
    gemm(qb + (long)b * SEQ * DM, kb + (long)b * SEQ * DM, nullptr, Sws, nullptr,
         (int)SEQ, (int)SEQ, 64, 1024, 1024, 1024, 64, 64, SEQ * SEQ, HB, iscl, 0, 0);
    softmax_row<<<dim3(HB * (unsigned)SEQ), dim3(256), 0, stream>>>(Sws, Pws, (int)SEQ);
    gemm(Pws, vtb + (long)b * HB * 64 * SEQ, nullptr, ctx + (long)b * SEQ * DM, nullptr,
         (int)SEQ, 64, (int)SEQ, 1024, 1024, 1024, SEQ * SEQ, 64 * SEQ, 64, HB, 1.0f, 0, 0);
  }
  add_ln<<<dim3((unsigned)TOK), dim3(256), 0, stream>>>(x1, ctx, g2, be2, x2, x2b);

  // ================= feed-forward =================
  gemm(x2b, w1b, b1, nullptr, hb, (int)TOK, (int)FFD, (int)DM, 1024, 1024, 4096, 0, 0, 0, 1, 1.0f, 1, 0);
  gemm(hb,  w2b, b2, ctx, nullptr, (int)TOK, (int)DM, (int)FFD, 4096, 4096, 1024, 0, 0, 0, 1, 1.0f, 0, 0);
  add_ln<<<dim3((unsigned)TOK), dim3(256), 0, stream>>>(x2, ctx, g3, be3, out, nullptr);
}
